// UNet4D_61426622267627
// MI455X (gfx1250) — matrix operand order repaired and checked by reading, not compiled, not run on hardware
//
#include <hip/hip_runtime.h>
#include <hip/hip_bf16.h>

typedef __attribute__((ext_vector_type(16))) _Float16 v16h;
typedef __attribute__((ext_vector_type(8)))  _Float16 v8h;
typedef __attribute__((ext_vector_type(8)))  float    v8f;

#define WPB 4  // waves per block (wave32 -> 128 threads)

// split-f16 planes: value ~= hi + lo, hi = f16(v), lo = f16(v - hi)
__device__ __forceinline__ void split2(float v, _Float16& h, _Float16& l) {
    h = (_Float16)v;
    l = (_Float16)(v - (float)h);
}

union U16H { v16h v; v8h g[2]; unsigned u[8]; };

struct CPd {
    int it, iz, iy, ixd;   // input dims
    int inCst;             // input channel stride (NHWC innermost size)
    int oz, oy, oxd;       // output dims (ot implied)
    int outCst;            // output channel stride
    int nMT;               // M tiles (16 ch)
    int total;             // nMT * nNT
    int hasBias;
};

// Implicit-GEMM 4D conv (normal / strided / transposed-as-dilated) on NHWC
// split-plane tensors. K ordering channel-inner: K = off*CIN + ci.
// D = A(hi)B(hi) + A(hi)B(lo) + A(lo)B(hi): ~f32 accuracy, f32 accumulate.
template<int CIN, int KT, int KZ, int KY, int KX,
         int PT, int PZ, int PY, int PX, int STRIDE, int UP>
__global__ __launch_bounds__(32 * WPB)
void conv4d_wmma(const _Float16* __restrict__ Xh, const _Float16* __restrict__ Xl,
                 const _Float16* __restrict__ Wh, const _Float16* __restrict__ Wl,
                 const float* __restrict__ Bv,
                 _Float16* __restrict__ Yh, _Float16* __restrict__ Yl, CPd p)
{
    constexpr int KV   = KT * KZ * KY * KX;
    constexpr int Ktot = CIN * KV;
    constexpr int LG   = (CIN == 1 ? 0 : CIN == 16 ? 4 : CIN == 32 ? 5 : CIN == 64 ? 6 : 7);
    constexpr bool TAIL = (Ktot % 32) != 0;

    const int wave = threadIdx.x >> 5;
    const int lane = threadIdx.x & 31;
    const int tile = blockIdx.x * WPB + wave;
    if (tile >= p.total) return;            // wave-uniform: EXEC all-1s at WMMA

    const int mt = tile % p.nMT;
    const int nt = tile / p.nMT;

    int otc[2], ozc[2], oyc[2], oxc[2], ncol[2];
#pragma unroll
    for (int t = 0; t < 2; ++t) {
        const int n = nt * 32 + t * 16 + (lane & 15);
        ncol[t] = n;
        const int t1 = n / p.oxd;  oxc[t] = n  - t1 * p.oxd;
        const int t2 = t1 / p.oy;  oyc[t] = t1 - t2 * p.oy;
        otc[t] = t2 / p.oz;        ozc[t] = t2 - otc[t] * p.oz;
    }

    const int co    = mt * 16 + (lane & 15);
    const int kofsA = (lane >> 4) * 8;
    const int kofsB = (lane >> 4) * 16;

    const v8h  z8  = {};
    const v16h z16 = {};

    v8f acc[2] = {{0,0,0,0,0,0,0,0}, {0,0,0,0,0,0,0,0}};

    for (int k0 = 0; k0 < Ktot; k0 += 32) {
        U16H ah, al;

        // ========= A fragment: weight planes [co][off][ci], 8 halfs/group =========
        if constexpr (CIN >= 16) {
#pragma unroll
            for (int g = 0; g < 2; ++g) {
                const int kb  = k0 + kofsB + g * 8;
                const int off = kb >> LG;                      // group-uniform
                const int cib = kb & (CIN - 1);
                const bool offok = !TAIL || (off < KV);        // uniform per lane half
                const int wb = (co * KV + off) * CIN + cib;    // 8 contiguous halfs
                ah.g[g] = offok ? *(const v8h*)(Wh + wb) : z8;
                al.g[g] = offok ? *(const v8h*)(Wl + wb) : z8;
            }
        } else {  // CIN == 1 (first conv only)
#pragma unroll
            for (int e = 0; e < 16; ++e) {
                const int v = e >> 1, h = e & 1;
                const int kk = k0 + kofsB + e;
                const bool okk = (!TAIL || kk < Ktot);
                ah.v[e] = okk ? Wh[co * KV + kk] : (_Float16)0;
                al.v[e] = okk ? Wl[co * KV + kk] : (_Float16)0;
            }
        }

        // ============ B fragments (NHWC patch planes) + WMMAs ============
#pragma unroll
        for (int t = 0; t < 2; ++t) {
            U16H bh, bl;
            if constexpr (CIN >= 16) {
                const int kb  = k0 + kofsB;
                const int off = kb >> LG;            // shared by all 16 elems
                const int cib = kb & (CIN - 1);
                int rem = off;
                const int at = rem / (KZ*KY*KX); rem -= at * (KZ*KY*KX);
                const int az = rem / (KY*KX);    rem -= az * (KY*KX);
                const int ay = rem / KX;         const int ax = rem - ay * KX;
                const int n0 = otc[t] * STRIDE - PT + at;
                const int n1 = ozc[t] * STRIDE - PZ + az;
                const int n2 = oyc[t] * STRIDE - PY + ay;
                const int n3 = oxc[t] * STRIDE - PX + ax;
                bool ok; int i0, i1, i2, i3;
                if constexpr (UP > 1) {
                    ok = ((n0 | n1 | n2 | n3) >= 0) &&
                         !((n0 & 1) | (n1 & 1) | (n2 & 1) | (n3 & 1));
                    i0 = n0 >> 1; i1 = n1 >> 1; i2 = n2 >> 1; i3 = n3 >> 1;
                } else {
                    ok = ((n0 | n1 | n2 | n3) >= 0);
                    i0 = n0; i1 = n1; i2 = n2; i3 = n3;
                }
                ok = ok && (i0 < p.it) && (i1 < p.iz) && (i2 < p.iy) && (i3 < p.ixd);
                if constexpr (TAIL) ok = ok && (off < KV);
                const int sp = ((i0 * p.iz + i1) * p.iy + i2) * p.ixd + i3;
                const int xb = sp * p.inCst + cib;   // 16 contiguous halfs (32B aligned)
                bh.v = ok ? *(const v16h*)(Xh + xb) : z16;
                bl.v = ok ? *(const v16h*)(Xl + xb) : z16;
            } else {  // CIN == 1
#pragma unroll
                for (int e = 0; e < 16; ++e) {
                    const int kk = k0 + kofsB + e;
                    _Float16 hh = (_Float16)0, ll = (_Float16)0;
                    if (!TAIL || kk < Ktot) {
                        int rem = kk;
                        const int at = rem / (KZ*KY*KX); rem -= at * (KZ*KY*KX);
                        const int az = rem / (KY*KX);    rem -= az * (KY*KX);
                        const int ay = rem / KX;         const int ax = rem - ay * KX;
                        const int n0 = otc[t] * STRIDE - PT + at;
                        const int n1 = ozc[t] * STRIDE - PZ + az;
                        const int n2 = oyc[t] * STRIDE - PY + ay;
                        const int n3 = oxc[t] * STRIDE - PX + ax;
                        const bool ok = ((n0 | n1 | n2 | n3) >= 0) &&
                                        (n0 < p.it) && (n1 < p.iz) &&
                                        (n2 < p.iy) && (n3 < p.ixd);
                        if (ok) {
                            const int sp = ((n0 * p.iz + n1) * p.iy + n2) * p.ixd + n3;
                            hh = Xh[sp]; ll = Xl[sp];
                        }
                    }
                    bh.v[e] = hh; bl.v[e] = ll;
                }
            }
            acc[t] = __builtin_amdgcn_wmma_f32_16x16x32_f16(false, ah.v, false, bh.v,
                                                            (short)0, acc[t], false, false);
            acc[t] = __builtin_amdgcn_wmma_f32_16x16x32_f16(false, ah.v, false, bl.v,
                                                            (short)0, acc[t], false, false);
            acc[t] = __builtin_amdgcn_wmma_f32_16x16x32_f16(false, al.v, false, bh.v,
                                                            (short)0, acc[t], false, false);
        }
    }

    // ---- store planes (8 consecutive channels per lane half -> b128 per plane) ----
    const int mrow = (lane >> 4) * 8;
#pragma unroll
    for (int t = 0; t < 2; ++t) {
#pragma unroll
        for (int r = 0; r < 8; ++r) {
            const int c_ = mt * 16 + mrow + r;
            float v = acc[t][r];
            if (p.hasBias) v += Bv[c_];
            _Float16 h, l; split2(v, h, l);
            const int a = ncol[t] * p.outCst + c_;
            Yh[a] = h; Yl[a] = l;
        }
    }
}

// ---------------- repack / format-conversion kernels ----------------

__global__ __launch_bounds__(256)
void pack_tensor(const float* __restrict__ src, _Float16* __restrict__ dh,
                 _Float16* __restrict__ dl, int n)
{
    const int i = blockIdx.x * blockDim.x + threadIdx.x;
    if (i < n) { _Float16 h, l; split2(src[i], h, l); dh[i] = h; dl[i] = l; }
}

// [Cout][Cin][KV] f32 -> [Cout][KV][Cin] split planes
__global__ __launch_bounds__(256)
void repack_w(const float* __restrict__ src, _Float16* __restrict__ dh,
              _Float16* __restrict__ dl, int Cout, int Cin, int KV)
{
    const int i = blockIdx.x * blockDim.x + threadIdx.x;
    if (i >= Cout * Cin * KV) return;
    const int ci  = i % Cin;
    const int r   = i / Cin;
    const int off = r % KV;
    const int co  = r / KV;
    _Float16 h, l; split2(src[(co * Cin + ci) * KV + off], h, l);
    dh[i] = h; dl[i] = l;
}

// transposed conv: [Cin][Cout][KV] f32, flipped -> [Cout][KV][Cin] split planes
__global__ __launch_bounds__(256)
void repack_wT(const float* __restrict__ src, _Float16* __restrict__ dh,
               _Float16* __restrict__ dl, int Cout, int Cin,
               int KT, int KZ, int KY, int KX)
{
    const int KV = KT * KZ * KY * KX;
    const int i = blockIdx.x * blockDim.x + threadIdx.x;
    if (i >= Cout * Cin * KV) return;
    const int ci  = i % Cin;
    const int r   = i / Cin;
    const int off = r % KV;
    const int co  = r / KV;
    int rem = off;
    const int at = rem / (KZ*KY*KX); rem -= at * (KZ*KY*KX);
    const int az = rem / (KY*KX);    rem -= az * (KY*KX);
    const int ay = rem / KX;         const int ax = rem - ay * KX;
    const int offf = (((KT-1-at)*KZ + (KZ-1-az))*KY + (KY-1-ay))*KX + (KX-1-ax);
    _Float16 h, l; split2(src[(ci * Cout + co) * KV + offf], h, l);
    dh[i] = h; dl[i] = l;
}

// ---------------- GroupNorm on split-plane NHWC sub-tensors ----------------

__global__ __launch_bounds__(512)
void gn_stats(const _Float16* __restrict__ Xh, const _Float16* __restrict__ Xl,
              float* __restrict__ stats, int S, int Cst, int base, int cpg, int lgcpg)
{
    const int g = blockIdx.x;
    const int cb = base + g * cpg;
    const int count = S * cpg;
    float s = 0.f, s2 = 0.f;
    for (int i = threadIdx.x; i < count; i += blockDim.x) {
        const int sp = i >> lgcpg;
        const int j  = i & (cpg - 1);
        const int a  = sp * Cst + cb + j;
        const float v = (float)Xh[a] + (float)Xl[a];
        s += v; s2 += v * v;
    }
    __shared__ float sh[512], sh2[512];
    sh[threadIdx.x] = s; sh2[threadIdx.x] = s2;
    __syncthreads();
    for (int ofs = blockDim.x >> 1; ofs > 0; ofs >>= 1) {
        if ((int)threadIdx.x < ofs) {
            sh[threadIdx.x]  += sh[threadIdx.x + ofs];
            sh2[threadIdx.x] += sh2[threadIdx.x + ofs];
        }
        __syncthreads();
    }
    if (threadIdx.x == 0) {
        const float inv_n = 1.f / (float)count;
        const float m   = sh[0] * inv_n;
        const float var = sh2[0] * inv_n - m * m;    // biased var (torch GroupNorm)
        stats[2 * g]     = m;
        stats[2 * g + 1] = rsqrtf(var + 1e-5f);
    }
}

__global__ __launch_bounds__(256)
void gn_apply_lrelu(_Float16* __restrict__ Xh, _Float16* __restrict__ Xl,
                    const float* __restrict__ gamma, const float* __restrict__ beta,
                    const float* __restrict__ stats,
                    int S, int C, int lgC, int Cst, int base, int lgcpg)
{
    const int total = S * C;
    for (int i = blockIdx.x * blockDim.x + threadIdx.x; i < total;
         i += gridDim.x * blockDim.x) {
        const int sp = i >> lgC;
        const int c  = i & (C - 1);
        const int g  = c >> lgcpg;
        const int a  = sp * Cst + base + c;
        float v = (float)Xh[a] + (float)Xl[a];
        v = (v - stats[2 * g]) * stats[2 * g + 1];
        v = v * gamma[c] + beta[c];
        v = (v >= 0.f) ? v : 0.01f * v;
        _Float16 h, l; split2(v, h, l);
        Xh[a] = h; Xl[a] = l;
    }
}

// Final 1x1x1x1 conv: 16 -> 1 channel, split planes in, f32 out.
__global__ __launch_bounds__(256)
void conv1x1_out(const _Float16* __restrict__ Xh, const _Float16* __restrict__ Xl,
                 const float* __restrict__ W, const float* __restrict__ B,
                 float* __restrict__ Y, int S, int Cst)
{
    const int i = blockIdx.x * blockDim.x + threadIdx.x;
    if (i >= S) return;
    float s = B[0];
#pragma unroll
    for (int c = 0; c < 16; ++c)
        s += W[c] * ((float)Xh[i * Cst + c] + (float)Xl[i * Cst + c]);
    Y[i] = s;
}

// ---------------- host-side schedule ----------------

struct PT2 { _Float16* hi; _Float16* lo; };

template<int CIN, int KT, int KZ, int KY, int KX,
         int PT, int PZ, int PY, int PX, int STRIDE, int UP>
static void launch_conv(hipStream_t st, PT2 X, int inBase, int inCst,
                        PT2 Wt, const float* Bv,
                        PT2 Yt, int outBase, int outCst, int Cout,
                        const int id[4], const int od[4])
{
    CPd p;
    p.it = id[0]; p.iz = id[1]; p.iy = id[2]; p.ixd = id[3];
    p.inCst = inCst;
    p.oz = od[1]; p.oy = od[2]; p.oxd = od[3];
    p.outCst = outCst;
    const int Sout = od[0] * od[1] * od[2] * od[3];
    p.nMT = Cout / 16;
    p.total = p.nMT * (Sout / 32);
    p.hasBias = (Bv != nullptr);
    const int blocks = (p.total + WPB - 1) / WPB;
    conv4d_wmma<CIN,KT,KZ,KY,KX,PT,PZ,PY,PX,STRIDE,UP>
        <<<blocks, 32 * WPB, 0, st>>>(X.hi + inBase, X.lo + inBase,
                                      Wt.hi, Wt.lo, Bv,
                                      Yt.hi + outBase, Yt.lo + outBase, p);
}

static int ilog2(int v) { int l = 0; while ((1 << l) < v) ++l; return l; }

static void launch_gn(hipStream_t st, PT2 X, const float* g, const float* b,
                      float* stats, int C, int S, int Cst, int base)
{
    const int cpg = C / 8;
    gn_stats<<<8, 512, 0, st>>>(X.hi, X.lo, stats, S, Cst, base, cpg, ilog2(cpg));
    int blocks = (S * C + 255) / 256;
    if (blocks > 4096) blocks = 4096;
    gn_apply_lrelu<<<blocks, 256, 0, st>>>(X.hi, X.lo, g, b, stats, S, C, ilog2(C),
                                           Cst, base, ilog2(cpg));
}

extern "C" void kernel_launch(void* const* d_in, const int* in_sizes, int n_in,
                              void* d_out, int out_size, void* d_ws, size_t ws_size,
                              hipStream_t stream)
{
    (void)in_sizes; (void)n_in; (void)out_size; (void)ws_size;
    #define PF(i) ((const float*)d_in[i])
    const float* x = PF(0);
    const int C1 = 1, C2 = 7, C3 = 13, C4 = 19, U1C = 25, U2C = 31, U3C = 37;
    const int DS1W = 43, DS1B = 44, DS2W = 45, DS2B = 46, DS3W = 47, DS3B = 48;
    const int U1W = 49, U1B = 50, U2W = 51, U2B = 52, U3W = 53, U3B = 54;
    const int OW = 55, OB = 56;

    static const int L0[4] = {12, 32, 32, 32};
    static const int L1[4] = { 6, 16, 16, 16};
    static const int L2[4] = { 3,  8,  8,  8};
    static const int L3[4] = { 1,  4,  4,  4};

    const int S0 = 393216, S1 = 24576, S2 = 1536, S3 = 64;

    unsigned* ws = (unsigned*)d_ws;
    size_t off = 0;
    // n dwords -> hi plane (n halfs) + lo plane (n halfs)
    auto allocPT = [&](size_t n) {
        unsigned* b = ws + off; off += n;
        PT2 t; t.hi = (_Float16*)b; t.lo = (_Float16*)b + n; return t;
    };

    // activations (split-plane NHWC)
    PT2 xp    = allocPT((size_t)1 * S0);
    PT2 c1tmp = allocPT((size_t)16 * S0);
    PT2 cat3  = allocPT((size_t)32 * S0);   // ch 0-15: d1, ch 16-31: u3/h1
    PT2 ds1o  = allocPT((size_t)16 * S1);
    PT2 c2tmp = allocPT((size_t)32 * S1);
    PT2 cat2  = allocPT((size_t)64 * S1);   // ch 0-31: d3, ch 32-63: u2/h3
    PT2 ds2o  = allocPT((size_t)32 * S2);
    PT2 c3tmp = allocPT((size_t)64 * S2);
    PT2 cat1  = allocPT((size_t)128 * S2);  // ch 0-63: d5, ch 64-127: u1/h5
    PT2 ds3o  = allocPT((size_t)64 * S3);
    PT2 c4tmp = allocPT((size_t)128 * S3);
    PT2 d7    = allocPT((size_t)128 * S3);
    float* stats = (float*)(ws + off); off += 64;

    // repacked split-plane weights
    auto rp = [&](int wi, int Cout, int Cin, int KV) {
        PT2 dst = allocPT((size_t)Cout * Cin * KV);
        const int n = Cout * Cin * KV;
        repack_w<<<(n + 255) / 256, 256, 0, stream>>>(PF(wi), dst.hi, dst.lo, Cout, Cin, KV);
        return dst;
    };
    auto rpT = [&](int wi, int Cout, int Cin, int kt, int kz, int ky, int kx) {
        const int KV = kt * kz * ky * kx;
        PT2 dst = allocPT((size_t)Cout * Cin * KV);
        const int n = Cout * Cin * KV;
        repack_wT<<<(n + 255) / 256, 256, 0, stream>>>(PF(wi), dst.hi, dst.lo,
                                                       Cout, Cin, kt, kz, ky, kx);
        return dst;
    };
    PT2 w_c1a = rp(C1+0,  16,   1, 81);
    PT2 w_c1b = rp(C1+1,  16,  16, 81);
    PT2 w_c2a = rp(C2+0,  32,  16, 81);
    PT2 w_c2b = rp(C2+1,  32,  32, 81);
    PT2 w_c3a = rp(C3+0,  64,  32, 81);
    PT2 w_c3b = rp(C3+1,  64,  64, 81);
    PT2 w_c4a = rp(C4+0, 128,  64, 81);
    PT2 w_c4b = rp(C4+1, 128, 128, 81);
    PT2 w_u1a = rp(U1C+0, 64, 128, 81);
    PT2 w_u1b = rp(U1C+1, 64,  64, 81);
    PT2 w_u2a = rp(U2C+0, 32,  64, 81);
    PT2 w_u2b = rp(U2C+1, 32,  32, 81);
    PT2 w_u3a = rp(U3C+0, 16,  32, 81);
    PT2 w_u3b = rp(U3C+1, 16,  16, 81);
    PT2 w_ds1 = rp(DS1W,  16,  16, 16);
    PT2 w_ds2 = rp(DS2W,  32,  32, 16);
    PT2 w_ds3 = rp(DS3W,  64,  64, 16);
    PT2 w_u1T = rpT(U1W,  64, 128, 5, 4, 4, 4);
    PT2 w_u2T = rpT(U2W,  32,  64, 4, 4, 4, 4);
    PT2 w_u3T = rpT(U3W,  16,  32, 4, 4, 4, 4);

    pack_tensor<<<(S0 + 255) / 256, 256, 0, stream>>>(x, xp.hi, xp.lo, S0);

    // ---- encoder ----
    launch_conv<  1,3,3,3,3, 1,1,1,1, 1,1>(stream, xp, 0, 1,  w_c1a, nullptr, c1tmp, 0, 16, 16, L0, L0);
    launch_gn(stream, c1tmp, PF(C1+2), PF(C1+3), stats, 16, S0, 16, 0);
    launch_conv< 16,3,3,3,3, 1,1,1,1, 1,1>(stream, c1tmp, 0, 16, w_c1b, nullptr, cat3, 0, 32, 16, L0, L0); // d1
    launch_gn(stream, cat3, PF(C1+4), PF(C1+5), stats, 16, S0, 32, 0);

    launch_conv< 16,2,2,2,2, 0,0,0,0, 2,1>(stream, cat3, 0, 32, w_ds1, PF(DS1B), ds1o, 0, 16, 16, L0, L1);
    launch_conv< 16,3,3,3,3, 1,1,1,1, 1,1>(stream, ds1o, 0, 16, w_c2a, nullptr, c2tmp, 0, 32, 32, L1, L1);
    launch_gn(stream, c2tmp, PF(C2+2), PF(C2+3), stats, 32, S1, 32, 0);
    launch_conv< 32,3,3,3,3, 1,1,1,1, 1,1>(stream, c2tmp, 0, 32, w_c2b, nullptr, cat2, 0, 64, 32, L1, L1); // d3
    launch_gn(stream, cat2, PF(C2+4), PF(C2+5), stats, 32, S1, 64, 0);

    launch_conv< 32,2,2,2,2, 0,0,0,0, 2,1>(stream, cat2, 0, 64, w_ds2, PF(DS2B), ds2o, 0, 32, 32, L1, L2);
    launch_conv< 32,3,3,3,3, 1,1,1,1, 1,1>(stream, ds2o, 0, 32, w_c3a, nullptr, c3tmp, 0, 64, 64, L2, L2);
    launch_gn(stream, c3tmp, PF(C3+2), PF(C3+3), stats, 64, S2, 64, 0);
    launch_conv< 64,3,3,3,3, 1,1,1,1, 1,1>(stream, c3tmp, 0, 64, w_c3b, nullptr, cat1, 0, 128, 64, L2, L2); // d5
    launch_gn(stream, cat1, PF(C3+4), PF(C3+5), stats, 64, S2, 128, 0);

    launch_conv< 64,2,2,2,2, 0,0,0,0, 2,1>(stream, cat1, 0, 128, w_ds3, PF(DS3B), ds3o, 0, 64, 64, L2, L3);
    launch_conv< 64,3,3,3,3, 1,1,1,1, 1,1>(stream, ds3o, 0, 64, w_c4a, nullptr, c4tmp, 0, 128, 128, L3, L3);
    launch_gn(stream, c4tmp, PF(C4+2), PF(C4+3), stats, 128, S3, 128, 0);
    launch_conv<128,3,3,3,3, 1,1,1,1, 1,1>(stream, c4tmp, 0, 128, w_c4b, nullptr, d7, 0, 128, 128, L3, L3);
    launch_gn(stream, d7, PF(C4+4), PF(C4+5), stats, 128, S3, 128, 0);

    // ---- decoder ----
    launch_conv<128,5,4,4,4, 3,2,2,2, 1,2>(stream, d7, 0, 128, w_u1T, PF(U1B), cat1, 64, 128, 64, L3, L2);
    launch_conv<128,3,3,3,3, 1,1,1,1, 1,1>(stream, cat1, 0, 128, w_u1a, nullptr, c3tmp, 0, 64, 64, L2, L2);
    launch_gn(stream, c3tmp, PF(U1C+2), PF(U1C+3), stats, 64, S2, 64, 0);
    launch_conv< 64,3,3,3,3, 1,1,1,1, 1,1>(stream, c3tmp, 0, 64, w_u1b, nullptr, cat1, 64, 128, 64, L2, L2); // h5
    launch_gn(stream, cat1, PF(U1C+4), PF(U1C+5), stats, 64, S2, 128, 64);

    launch_conv< 64,4,4,4,4, 2,2,2,2, 1,2>(stream, cat1, 64, 128, w_u2T, PF(U2B), cat2, 32, 64, 32, L2, L1);
    launch_conv< 64,3,3,3,3, 1,1,1,1, 1,1>(stream, cat2, 0, 64, w_u2a, nullptr, c2tmp, 0, 32, 32, L1, L1);
    launch_gn(stream, c2tmp, PF(U2C+2), PF(U2C+3), stats, 32, S1, 32, 0);
    launch_conv< 32,3,3,3,3, 1,1,1,1, 1,1>(stream, c2tmp, 0, 32, w_u2b, nullptr, cat2, 32, 64, 32, L1, L1); // h3
    launch_gn(stream, cat2, PF(U2C+4), PF(U2C+5), stats, 32, S1, 64, 32);

    launch_conv< 32,4,4,4,4, 2,2,2,2, 1,2>(stream, cat2, 32, 64, w_u3T, PF(U3B), cat3, 16, 32, 16, L1, L0);
    launch_conv< 32,3,3,3,3, 1,1,1,1, 1,1>(stream, cat3, 0, 32, w_u3a, nullptr, c1tmp, 0, 16, 16, L0, L0);
    launch_gn(stream, c1tmp, PF(U3C+2), PF(U3C+3), stats, 16, S0, 16, 0);
    launch_conv< 16,3,3,3,3, 1,1,1,1, 1,1>(stream, c1tmp, 0, 16, w_u3b, nullptr, cat3, 16, 32, 16, L0, L0); // h1
    launch_gn(stream, cat3, PF(U3C+4), PF(U3C+5), stats, 16, S0, 32, 16);

    conv1x1_out<<<(S0 + 255) / 256, 256, 0, stream>>>(cat3.hi + 16, cat3.lo + 16,
                                                      PF(OW), PF(OB),
                                                      (float*)d_out, S0, 32);
    #undef PF
}
